// SparseResBlock_22728966930602
// MI455X (gfx1250) — matrix operand order repaired and checked by reading, not compiled, not run on hardware
//
#include <hip/hip_runtime.h>

#define NPTS   1000000
#define CCH    64
#define NTILES (NPTS/16)      // 62500 row-tiles of 16
#define EPSV   1e-5f
#define NELT4  (NPTS*CCH/4)   // 16,000,000 float4-granules

typedef __bf16 bf16;
typedef bf16  v16bf __attribute__((ext_vector_type(16)));
typedef float v8f   __attribute__((ext_vector_type(8)));

union AB32 { uint4 u[2]; v16bf v; };          // 32-byte WMMA operand fragment

// ---------------------------------------------------------------- prep: f32 -> bf16 features
__global__ void prep_feat(const float* __restrict__ f, bf16* __restrict__ fb) {
  size_t i = (size_t)blockIdx.x * blockDim.x + threadIdx.x;
  if (i >= (size_t)NELT4) return;
  float4 v = ((const float4*)f)[i];
  union { bf16 b[4]; uint2 u; } o;
  o.b[0] = (bf16)v.x; o.b[1] = (bf16)v.y; o.b[2] = (bf16)v.z; o.b[3] = (bf16)v.w;
  ((uint2*)fb)[i] = o.u;
}

// ---------------------------------------------------------------- prep: W[9,64,64] f32 -> transposed bf16 Wt[9][n][c]; zero stats
__global__ void prep_w(const float* __restrict__ W1, const float* __restrict__ W2,
                       bf16* __restrict__ wt1, bf16* __restrict__ wt2,
                       float* __restrict__ stats) {
  int i = blockIdx.x * blockDim.x + threadIdx.x;
  if (i < 256) stats[i] = 0.0f;               // sum1/sq1/sum2/sq2, 64 each
  if (i >= 2 * 9 * 64 * 64) return;
  int which = (i >= 9 * 64 * 64);
  int j = i - which * 9 * 64 * 64;
  int k = j >> 12, c = (j >> 6) & 63, n = j & 63;
  const float* src = which ? W2 : W1;
  bf16* dst = which ? wt2 : wt1;
  dst[k * 4096 + n * 64 + c] = (bf16)src[j];  // B-matrix friendly: K contiguous per column n
}

// ---------------------------------------------------------------- main conv: out[N,64] = sum_k gather(fin, nbr[:,k]) @ W[k]
// Also accumulates per-channel sum / sum-of-squares for the following batchnorm.
__global__ __launch_bounds__(256)
void subm_conv_wmma(const bf16* __restrict__ fin, const int* __restrict__ nbr,
                    const bf16* __restrict__ wt, float* __restrict__ out,
                    float* __restrict__ gsum, float* __restrict__ gsq) {
  extern __shared__ char smemraw[];
  bf16*  Wl    = (bf16*)smemraw;                        // 9*4096 bf16 = 73728 B, transposed W
  float* sstat = (float*)(smemraw + 9 * 4096 * 2);      // 128 floats

  // stage transposed bf16 weights into LDS (b128 transfers)
  for (int i = threadIdx.x; i < 9 * 4096 / 8; i += 256)
    ((uint4*)Wl)[i] = ((const uint4*)wt)[i];
  if (threadIdx.x < 128) sstat[threadIdx.x] = 0.0f;
  __syncthreads();

  const int lane = threadIdx.x & 31;
  const int wv   = threadIdx.x >> 5;
  const int hf   = lane >> 4;       // half-wave: selects K-subblock per ISA A/B layouts
  const int l16  = lane & 15;

  float ssum[4] = {0.f, 0.f, 0.f, 0.f};
  float ssq [4] = {0.f, 0.f, 0.f, 0.f};

  for (int tile = blockIdx.x * 8 + wv; tile < NTILES; tile += gridDim.x * 8) {
    const int row = tile * 16 + l16;           // A-matrix row for this lane

    // hoist all 9 neighbor indices (contiguous) and prefetch the gathered rows
    int idxs[9];
    #pragma unroll
    for (int k = 0; k < 9; ++k) idxs[k] = nbr[row * 9 + k];
    #pragma unroll
    for (int k = 0; k < 9; ++k) {
      const bf16* p = fin + (size_t)(idxs[k] < 0 ? 0 : idxs[k]) * CCH;
      __builtin_prefetch(p, 0, 1);             // global_prefetch_b8: warm the 128B row
    }

    v8f acc[4];
    #pragma unroll
    for (int t = 0; t < 4; ++t) acc[t] = (v8f)0.0f;

    #pragma unroll
    for (int k = 0; k < 9; ++k) {
      const int idx = idxs[k];
      const bool valid = idx >= 0;
      const uint4* src = (const uint4*)(fin + (size_t)(valid ? idx : 0) * CCH);
      const uint4 z = make_uint4(0u, 0u, 0u, 0u);
      // A (16x32 bf16): lanes 0-15 K{0..7,16..23}(+chunk), lanes 16-31 K{8..15,24..31}
      AB32 A0, A1;
      A0.u[0] = valid ? src[0 + hf] : z;    // K  0.. 7  (+8*hf)
      A0.u[1] = valid ? src[2 + hf] : z;    // K 16..23  (+8*hf)
      A1.u[0] = valid ? src[4 + hf] : z;    // K 32..39  (+8*hf)
      A1.u[1] = valid ? src[6 + hf] : z;    // K 48..55  (+8*hf)

      const bf16* wk = Wl + k * 4096;
      #pragma unroll
      for (int t = 0; t < 4; ++t) {         // 4 N-tiles of 16 output channels
        const int n = t * 16 + l16;
        // B (32x16 bf16): lane column n, K {0..7,16..23} + 8*hf, matching A
        const uint4* bp = (const uint4*)(wk + n * 64 + 8 * hf);
        AB32 B0, B1;
        B0.u[0] = bp[0]; B0.u[1] = bp[2];   // K chunk 0
        B1.u[0] = bp[4]; B1.u[1] = bp[6];   // K chunk 1 (+32 halves)
        acc[t] = __builtin_amdgcn_wmma_f32_16x16x32_bf16(
            false, A0.v, false, B0.v, (short)0, acc[t], false, false);
        acc[t] = __builtin_amdgcn_wmma_f32_16x16x32_bf16(
            false, A1.v, false, B1.v, (short)0, acc[t], false, false);
      }
    }

    // C/D layout: VGPR j -> row (j + 8*hf), column l16 within the n-tile
    const size_t obase = (size_t)(tile * 16 + hf * 8) * CCH;
    #pragma unroll
    for (int t = 0; t < 4; ++t) {
      const int col = t * 16 + l16;
      #pragma unroll
      for (int j = 0; j < 8; ++j) {
        float x = acc[t][j];
        out[obase + (size_t)j * CCH + col] = x;
        ssum[t] += x;
        ssq [t] += x * x;
      }
    }
  }

  // fold the two half-waves (same channel) then reduce across waves in LDS
  #pragma unroll
  for (int t = 0; t < 4; ++t) {
    ssum[t] += __shfl_xor(ssum[t], 16, 32);
    ssq [t] += __shfl_xor(ssq [t], 16, 32);
  }
  if (hf == 0) {
    #pragma unroll
    for (int t = 0; t < 4; ++t) {
      const int c = t * 16 + l16;
      atomicAdd(&sstat[c],      ssum[t]);
      atomicAdd(&sstat[64 + c], ssq [t]);
    }
  }
  __syncthreads();
  if (threadIdx.x < 64)        atomicAdd(&gsum[threadIdx.x],      sstat[threadIdx.x]);
  else if (threadIdx.x < 128)  atomicAdd(&gsq [threadIdx.x - 64], sstat[threadIdx.x]);
}

// ---------------------------------------------------------------- BN + relu -> bf16 activations
__global__ void bn_relu_bf16(const float* __restrict__ x, const float* __restrict__ gsum,
                             const float* __restrict__ gsq, const float* __restrict__ gamma,
                             const float* __restrict__ beta, bf16* __restrict__ y) {
  size_t i = (size_t)blockIdx.x * blockDim.x + threadIdx.x;
  if (i >= (size_t)NELT4) return;
  const int c = (int)((i * 4) & 63);
  float4 v = ((const float4*)x)[i];
  float vv[4] = {v.x, v.y, v.z, v.w};
  union { bf16 b[4]; uint2 u; } o;
  #pragma unroll
  for (int j = 0; j < 4; ++j) {
    float m   = gsum[c + j] * (1.0f / NPTS);
    float var = gsq [c + j] * (1.0f / NPTS) - m * m;
    float inv = rsqrtf(var + EPSV);
    float r   = (vv[j] - m) * inv * gamma[c + j] + beta[c + j];
    o.b[j] = (bf16)fmaxf(r, 0.0f);
  }
  ((uint2*)y)[i] = o.u;
}

// ---------------------------------------------------------------- BN + residual + relu, in place on d_out
__global__ void bn_res_relu(float* __restrict__ io, const float* __restrict__ feat,
                            const float* __restrict__ gsum, const float* __restrict__ gsq,
                            const float* __restrict__ gamma, const float* __restrict__ beta) {
  size_t i = (size_t)blockIdx.x * blockDim.x + threadIdx.x;
  if (i >= (size_t)NELT4) return;
  const int c = (int)((i * 4) & 63);
  float4 v = ((float4*)io)[i];
  float4 f = ((const float4*)feat)[i];
  float vv[4] = {v.x, v.y, v.z, v.w};
  float ff[4] = {f.x, f.y, f.z, f.w};
  float rr[4];
  #pragma unroll
  for (int j = 0; j < 4; ++j) {
    float m   = gsum[c + j] * (1.0f / NPTS);
    float var = gsq [c + j] * (1.0f / NPTS) - m * m;
    float inv = rsqrtf(var + EPSV);
    rr[j] = fmaxf((vv[j] - m) * inv * gamma[c + j] + beta[c + j] + ff[j], 0.0f);
  }
  ((float4*)io)[i] = make_float4(rr[0], rr[1], rr[2], rr[3]);
}

// ---------------------------------------------------------------- host launch
extern "C" void kernel_launch(void* const* d_in, const int* in_sizes, int n_in,
                              void* d_out, int out_size, void* d_ws, size_t ws_size,
                              hipStream_t stream) {
  const float* feat = (const float*)d_in[0];
  const int*   nbr  = (const int*)  d_in[1];
  const float* W1   = (const float*)d_in[2];
  const float* g1   = (const float*)d_in[3];
  const float* b1   = (const float*)d_in[4];
  const float* W2   = (const float*)d_in[5];
  const float* g2   = (const float*)d_in[6];
  const float* b2   = (const float*)d_in[7];
  float* out = (float*)d_out;

  char* ws = (char*)d_ws;
  bf16*  fb    = (bf16*)ws;                              // 128,000,000 B  bf16 features
  bf16*  act1  = (bf16*)(ws + 128000000);                // 128,000,000 B  bf16 activations
  float* stats = (float*)(ws + 256000000);               // 256 floats: sum1,sq1,sum2,sq2
  bf16*  wt1   = (bf16*)(ws + 256001024);                // 73,728 B transposed bf16 W1
  bf16*  wt2   = (bf16*)(ws + 256001024 + 73728);        // 73,728 B transposed bf16 W2

  const size_t lds = 9 * 4096 * 2 + 128 * 4;             // 74,240 B (<320 KB WGP budget)
  const int eltBlocks = (NELT4 + 255) / 256;

  prep_feat<<<eltBlocks, 256, 0, stream>>>(feat, fb);
  prep_w<<<(2 * 9 * 4096 + 255) / 256, 256, 0, stream>>>(W1, W2, wt1, wt2, stats);

  subm_conv_wmma<<<2048, 256, lds, stream>>>(fb, nbr, wt1, out, stats + 0, stats + 64);
  bn_relu_bf16<<<eltBlocks, 256, 0, stream>>>(out, stats + 0, stats + 64, g1, b1, act1);
  subm_conv_wmma<<<2048, 256, lds, stream>>>(act1, nbr, wt2, out, stats + 128, stats + 192);
  bn_res_relu<<<eltBlocks, 256, 0, stream>>>(out, feat, stats + 128, stats + 192, g2, b2);
}
